// SelfAttention_27195732918937
// MI455X (gfx1250) — matrix operand order repaired and checked by reading, not compiled, not run on hardware
//
#include <hip/hip_runtime.h>
#include <hip/hip_bf16.h>

// ---------------------------------------------------------------------------
// Causal self-attention for MI455X (gfx1250), bf16 WMMA everywhere.
// B=4, S=2048, D=1024, H=16, Dh=64.
// GEMMs stage weight tiles into LDS with gfx1250 async-to-LDS copies
// (ASYNCcnt), double-buffered; WMMA consumes B fragments from LDS.
// ---------------------------------------------------------------------------

#define BATCH   4
#define SEQ     2048
#define DMODEL  1024
#define NHEADS  16
#define DHEAD   64

// B tile in LDS: 64 columns x 32 K-halves, padded to 40 halves (80B) per row
// so the 16B ds_load_b128 fragment reads spread across banks.
#define BROW    40
#define BTILE   (64 * BROW)

typedef __attribute__((ext_vector_type(16))) __bf16 bf16x16;
typedef __attribute__((ext_vector_type(8)))  __bf16 bf16x8;
typedef __attribute__((ext_vector_type(8)))  float  v8f;

// ---------------------------------------------------------------------------
// WMMA helper: D = A(16x32 bf16) * B(32x16 bf16) + C(16x16 f32)
// ---------------------------------------------------------------------------
__device__ inline v8f wmma_bf16(bf16x16 a, bf16x16 b, v8f c) {
  return __builtin_amdgcn_wmma_f32_16x16x32_bf16(
      /*neg_a=*/false, a, /*neg_b=*/false, b,
      /*c_mod=*/(short)0, c, /*reuse_a=*/false, /*reuse_b=*/false);
}

// A-fragment (16x32, M x K), row-major source with leading dim ld.
// Contiguous k order (same as B): lane L holds row M = row0 + (L%16); halves e=0..15 -> K = k0 +
// (L/16)*16 + e (one contiguous 32B run). Two 16B loads.
__device__ inline bf16x16 load_a_frag(const __bf16* base, int ld, int row0,
                                      int k0, int lane) {
  const __bf16* p = base + (long)(row0 + (lane & 15)) * ld + k0 + ((lane >> 4) * 16);
  bf16x8 lo = *(const bf16x8*)(p);
  bf16x8 hi = *(const bf16x8*)(p + 8);
  bf16x16 a;
#pragma unroll
  for (int i = 0; i < 8; ++i) { a[i] = lo[i]; a[8 + i] = hi[i]; }
  return a;
}

// B-fragment (32x16, K x N) from a global row-major source (each output col N
// stored as a contiguous row, i.e. B^T). lane L holds col N = n0 + (L%16);
// halves e=0..15 -> K = k0 + (L/16)*16 + e (one contiguous 32B run).
__device__ inline bf16x16 load_b_frag(const __bf16* base, int ld, int n0,
                                      int k0, int lane) {
  const __bf16* p = base + (long)(n0 + (lane & 15)) * ld + k0 + ((lane >> 4) * 16);
  bf16x8 lo = *(const bf16x8*)(p);
  bf16x8 hi = *(const bf16x8*)(p + 8);
  bf16x16 b;
#pragma unroll
  for (int i = 0; i < 8; ++i) { b[i] = lo[i]; b[8 + i] = hi[i]; }
  return b;
}

// Same fragment but from the padded LDS B tile (rows of BROW halves).
__device__ inline bf16x16 load_b_frag_lds(const __bf16* bs, int n0, int lane) {
  const __bf16* p = bs + (n0 + (lane & 15)) * BROW + ((lane >> 4) * 16);
  bf16x8 lo = *(const bf16x8*)(p);
  bf16x8 hi = *(const bf16x8*)(p + 8);
  bf16x16 b;
#pragma unroll
  for (int i = 0; i < 8; ++i) { b[i] = lo[i]; b[8 + i] = hi[i]; }
  return b;
}

// ---------------------------------------------------------------------------
// Async stage of one 64x32 B tile (weights) into LDS: 256 threads issue one
// global_load_async_to_lds_b128 each (thread t: col = t/4, 16B chunk = t%4).
// Tracked by ASYNCcnt; VDST operand carries the LDS byte address (low 32 bits
// of the generic shared pointer, per the ISA flat->LDS mapping).
// ---------------------------------------------------------------------------
__device__ inline void stage_b_async(const __bf16* w_mat, int col0,
                                     __bf16* bsm, int buf, int kk) {
  const int c    = threadIdx.x >> 2;  // 0..63 : output column within tile
  const int part = threadIdx.x & 3;   // 0..3  : 16B chunk within the K-row
  const __bf16* g = w_mat + (long)(col0 + c) * DMODEL + kk + part * 8;
  unsigned int lds = (unsigned int)(size_t)(bsm + buf * BTILE + c * BROW + part * 8);
  asm volatile("global_load_async_to_lds_b128 %0, %1, off"
               :: "v"(lds), "v"(g) : "memory");
}

__device__ inline void wait_async_all() {
  asm volatile("s_wait_asynccnt 0x0" ::: "memory");
}

// ---------------------------------------------------------------------------
// Shared GEMM core: wave-private 16x64 fp32 accumulator, A from global,
// B tile double-buffered in LDS via async copies.
// ---------------------------------------------------------------------------
__device__ inline void gemm_core_staged(const __bf16* a_mat, const __bf16* w_mat,
                                        int row0, int col0, __bf16* bsm,
                                        v8f acc[4]) {
  const int lane = threadIdx.x & 31;
  stage_b_async(w_mat, col0, bsm, 0, 0);
  for (int kk = 0; kk < DMODEL; kk += 32) {
    wait_async_all();      // my wave's pending tile loads are done
    __syncthreads();       // everyone's are done -> tile complete
    const int cur = (kk >> 5) & 1;
    if (kk + 32 < DMODEL) stage_b_async(w_mat, col0, bsm, cur ^ 1, kk + 32);
    __builtin_prefetch(a_mat + (long)(row0 + lane) * DMODEL + kk + 64, 0, 1);
    bf16x16 a = load_a_frag(a_mat, DMODEL, row0, kk, lane);
    const __bf16* bs = bsm + cur * BTILE;
#pragma unroll
    for (int n = 0; n < 4; ++n)
      acc[n] = wmma_bf16(a, load_b_frag_lds(bs, n * 16, lane), acc[n]);
    __syncthreads();       // readers done before this buffer is re-staged
  }
}

// ---------------------------------------------------------------------------
// Kernel 1: fp32 -> bf16 streaming convert
// ---------------------------------------------------------------------------
__global__ __launch_bounds__(256) void f32_to_bf16_kernel(const float* __restrict__ in,
                                                          __bf16* __restrict__ out,
                                                          int n) {
  int i = blockIdx.x * blockDim.x + threadIdx.x;
  int stride = gridDim.x * blockDim.x;
  for (; i < n; i += stride) out[i] = (__bf16)in[i];
}

// ---------------------------------------------------------------------------
// Kernel 2: QKV projection. qkv[r,e] = x[r,:]·W_in[e,:] + b_in[e]
// Block = 256 thr = 8 waves; block tile 128x64; wave tile 16x64.
// Epilogue scatters into Q/K [B,H,S,Dh] and V transposed [B,H,Dh,S] (bf16).
// ---------------------------------------------------------------------------
__global__ __launch_bounds__(256)
void qkv_gemm_kernel(const __bf16* __restrict__ xb, const __bf16* __restrict__ wb,
                     const float* __restrict__ bias, __bf16* __restrict__ qb,
                     __bf16* __restrict__ kb, __bf16* __restrict__ vb) {
  __shared__ __bf16 bsm[2 * BTILE];
  const int lane = threadIdx.x & 31;
  const int w    = threadIdx.x >> 5;
  const int ln   = lane & 15;
  const int half = lane >> 4;
  const int row0 = blockIdx.y * 128 + w * 16;
  const int col0 = blockIdx.x * 64;

  v8f acc[4] = {};
  gemm_core_staged(xb, wb, row0, col0, bsm, acc);

  const int chunk = col0 / DMODEL;            // 0=q 1=k 2=v
  const int h     = (col0 % DMODEL) / DHEAD;  // 64-wide tile => single head
#pragma unroll
  for (int n = 0; n < 4; ++n) {
    const int col = col0 + n * 16 + ln;
    const int dh  = (n * 16 + ln) & (DHEAD - 1);
    const float bv = bias[col];
#pragma unroll
    for (int r = 0; r < 8; ++r) {
      const int row = row0 + r + half * 8;     // C layout: VGPR r -> M = r + 8*(L/16)
      const int bidx = row / SEQ;
      const int s    = row % SEQ;
      const long bh  = (long)bidx * NHEADS + h;
      const __bf16 o = (__bf16)(acc[n][r] + bv);
      if (chunk == 0)      qb[(bh * SEQ + s) * DHEAD + dh] = o;
      else if (chunk == 1) kb[(bh * SEQ + s) * DHEAD + dh] = o;
      else                 vb[(bh * DHEAD + dh) * SEQ + s] = o;  // V transposed
    }
  }
}

// ---------------------------------------------------------------------------
// Kernel 3: flash attention. Grid (B*H, S/128); 8 waves; wave owns 16 q-rows.
// Per 32-key step: 4 QK WMMAs, online softmax, LDS relayout of P, 4 PV WMMAs.
// ---------------------------------------------------------------------------
__global__ __launch_bounds__(256)
void attn_kernel(const __bf16* __restrict__ qb, const __bf16* __restrict__ kb,
                 const __bf16* __restrict__ vb, __bf16* __restrict__ ob) {
  __shared__ __bf16 plds[8][16][32];  // per-wave 16x32 P tile (1 KB/wave)

  const int lane = threadIdx.x & 31;
  const int w    = threadIdx.x >> 5;
  const int ln   = lane & 15;
  const int half = lane >> 4;
  const int bh   = blockIdx.x;
  const int b    = bh >> 4;
  const int h    = bh & 15;
  const int q0   = blockIdx.y * 128 + w * 16;

  const __bf16* Q  = qb + (long)bh * SEQ * DHEAD;   // [S, Dh]
  const __bf16* Km = kb + (long)bh * SEQ * DHEAD;   // [S, Dh]
  const __bf16* Vt = vb + (long)bh * DHEAD * SEQ;   // [Dh, S] (transposed)

  // Q fragments, resident for the whole key loop (K-dim 64 -> 2 frags)
  const bf16x16 aq0 = load_a_frag(Q, DHEAD, q0, 0, lane);
  const bf16x16 aq1 = load_a_frag(Q, DHEAD, q0, 32, lane);

  v8f acc[4] = {};
  float m[8], l[8];
#pragma unroll
  for (int r = 0; r < 8; ++r) { m[r] = -3.0e38f; l[r] = 0.0f; }

  const float scl = 0.125f;  // 1/sqrt(Dh)
  for (int j = 0; j < q0 + 16; j += 32) {   // causal limit (wave-uniform)
    // ---- scores: two 16x16 tiles covering keys j..j+31
    v8f sc[2];
#pragma unroll
    for (int t = 0; t < 2; ++t) {
      v8f s = {};
      s = wmma_bf16(aq0, load_b_frag(Km, DHEAD, j + t * 16, 0, lane), s);
      s = wmma_bf16(aq1, load_b_frag(Km, DHEAD, j + t * 16, 32, lane), s);
      sc[t] = s;
    }
    // ---- scale + causal mask, row max
    float mx[8];
#pragma unroll
    for (int r = 0; r < 8; ++r) {
      const int qa = q0 + r + half * 8;
      float s0 = sc[0][r] * scl;
      float s1 = sc[1][r] * scl;
      if (j + ln > qa)      s0 = -3.0e38f;
      if (j + 16 + ln > qa) s1 = -3.0e38f;
      sc[0][r] = s0; sc[1][r] = s1;
      float v = fmaxf(s0, s1);
      v = fmaxf(v, __shfl_xor(v, 1, 32));
      v = fmaxf(v, __shfl_xor(v, 2, 32));
      v = fmaxf(v, __shfl_xor(v, 4, 32));
      v = fmaxf(v, __shfl_xor(v, 8, 32));   // reduce within 16-lane half
      mx[r] = v;
    }
    // ---- online softmax update
    float alpha[8];
#pragma unroll
    for (int r = 0; r < 8; ++r) {
      const float mn = fmaxf(m[r], mx[r]);
      alpha[r] = __expf(m[r] - mn);
      m[r] = mn;
      const float p0 = __expf(sc[0][r] - mn);
      const float p1 = __expf(sc[1][r] - mn);
      sc[0][r] = p0; sc[1][r] = p1;
      float sum = p0 + p1;
      sum += __shfl_xor(sum, 1, 32);
      sum += __shfl_xor(sum, 2, 32);
      sum += __shfl_xor(sum, 4, 32);
      sum += __shfl_xor(sum, 8, 32);
      l[r] = l[r] * alpha[r] + sum;
    }
#pragma unroll
    for (int n = 0; n < 4; ++n)
#pragma unroll
      for (int r = 0; r < 8; ++r) acc[n][r] *= alpha[r];

    // ---- P: C-layout -> A-layout via per-wave LDS tile (DS in-order per wave)
#pragma unroll
    for (int t = 0; t < 2; ++t)
#pragma unroll
      for (int r = 0; r < 8; ++r)
        plds[w][r + half * 8][t * 16 + ln] = (__bf16)sc[t][r];

    bf16x8 lo = *(const bf16x8*)&plds[w][ln][half * 16];
    bf16x8 hi = *(const bf16x8*)&plds[w][ln][half * 16 + 8];
    bf16x16 ap;
#pragma unroll
    for (int i = 0; i < 8; ++i) { ap[i] = lo[i]; ap[8 + i] = hi[i]; }

    // ---- out += P(16x32) @ V(32x64): 4 WMMAs, V^T rows are contiguous runs
#pragma unroll
    for (int n = 0; n < 4; ++n) {
      bf16x16 bv = load_b_frag(Vt, SEQ, n * 16, j, lane);
      acc[n] = wmma_bf16(ap, bv, acc[n]);
    }
  }

  // ---- normalize and store into [B,S,D] bf16 for the out-proj GEMM
#pragma unroll
  for (int r = 0; r < 8; ++r) {
    const float inv = 1.0f / l[r];
    const long row = (long)b * SEQ + q0 + r + half * 8;
#pragma unroll
    for (int n = 0; n < 4; ++n)
      ob[row * DMODEL + h * DHEAD + n * 16 + ln] = (__bf16)(acc[n][r] * inv);
  }
}

// ---------------------------------------------------------------------------
// Kernel 4: out projection. out[r,e] = attn[r,:]·W_out[e,:] + b_out[e]  (fp32)
// ---------------------------------------------------------------------------
__global__ __launch_bounds__(256)
void out_gemm_kernel(const __bf16* __restrict__ ab, const __bf16* __restrict__ wb,
                     const float* __restrict__ bias, float* __restrict__ out) {
  __shared__ __bf16 bsm[2 * BTILE];
  const int lane = threadIdx.x & 31;
  const int w    = threadIdx.x >> 5;
  const int ln   = lane & 15;
  const int half = lane >> 4;
  const int row0 = blockIdx.y * 128 + w * 16;
  const int col0 = blockIdx.x * 64;

  v8f acc[4] = {};
  gemm_core_staged(ab, wb, row0, col0, bsm, acc);

#pragma unroll
  for (int n = 0; n < 4; ++n) {
    const int col = col0 + n * 16 + ln;
    const float bv = bias[col];
#pragma unroll
    for (int r = 0; r < 8; ++r) {
      const long row = row0 + r + half * 8;
      out[row * DMODEL + col] = acc[n][r] + bv;
    }
  }
}

// ---------------------------------------------------------------------------
// Host launcher
// ---------------------------------------------------------------------------
extern "C" void kernel_launch(void* const* d_in, const int* in_sizes, int n_in,
                              void* d_out, int out_size, void* d_ws, size_t ws_size,
                              hipStream_t stream) {
  const float* x     = (const float*)d_in[0];  // [4,2048,1024]
  const float* W_in  = (const float*)d_in[1];  // [3072,1024]
  const float* b_in  = (const float*)d_in[2];  // [3072]
  const float* W_out = (const float*)d_in[3];  // [1024,1024]
  const float* b_out = (const float*)d_in[4];  // [1024]
  float* out = (float*)d_out;                  // [4,2048,1024]

  char* ws = (char*)d_ws;
  size_t off = 0;
  const size_t xN  = (size_t)BATCH * SEQ * DMODEL;       // 8,388,608
  const size_t wiN = (size_t)3 * DMODEL * DMODEL;        // 3,145,728
  const size_t woN = (size_t)DMODEL * DMODEL;            // 1,048,576
  const size_t hN  = (size_t)BATCH * NHEADS * SEQ * DHEAD;

  __bf16* xb  = (__bf16*)(ws + off); off += xN  * 2;
  __bf16* wib = (__bf16*)(ws + off); off += wiN * 2;
  __bf16* wob = (__bf16*)(ws + off); off += woN * 2;
  __bf16* qb  = (__bf16*)(ws + off); off += hN * 2;
  __bf16* kb  = (__bf16*)(ws + off); off += hN * 2;
  __bf16* vb  = (__bf16*)(ws + off); off += hN * 2;
  __bf16* ob  = (__bf16*)(ws + off); off += xN * 2;      // ~88 MB total

  f32_to_bf16_kernel<<<2048, 256, 0, stream>>>(x,     xb,  (int)xN);
  f32_to_bf16_kernel<<<2048, 256, 0, stream>>>(W_in,  wib, (int)wiN);
  f32_to_bf16_kernel<<<1024, 256, 0, stream>>>(W_out, wob, (int)woN);

  // QKV: M=8192 rows (64 blocks of 128), N=3072 (48 blocks of 64)
  qkv_gemm_kernel<<<dim3(48, 64), 256, 0, stream>>>(xb, wib, b_in, qb, kb, vb);

  // Attention: (B*H, S/128)
  attn_kernel<<<dim3(BATCH * NHEADS, SEQ / 128), 256, 0, stream>>>(qb, kb, vb, ob);

  // Out proj: N=1024 (16 blocks of 64), M=8192 (64 blocks of 128)
  out_gemm_kernel<<<dim3(16, 64), 256, 0, stream>>>(ob, wob, b_out, out);
}
